// VisionRetentionChunk_88845693485155
// MI455X (gfx1250) — matrix operand order repaired and checked by reading, not compiled, not run on hardware
//
#include <hip/hip_runtime.h>
#include <hip/hip_bf16.h>

// ---------------------------------------------------------------------------
// VisionRetentionChunk for MI455X (gfx1250, wave32, WMMA 16x16x32 bf16)
// Stages: weight bf16 convert -> QKV proj (WMMA) -> width attn (WMMA+softmax)
//         -> height attn (WMMA+softmax) -> lepe depthwise add (VALU)
//         -> out proj (WMMA)
// Tile staging uses GLOBAL_LOAD_ASYNC_TO_LDS_B128 (ASYNCcnt path).
// ---------------------------------------------------------------------------

#define CB   16
#define CH   56
#define CW   56
#define CC   256
#define NH   8
#define DK   32
#define TOK  (CB*CH*CW)          // 50176 tokens, divisible by 16

typedef __attribute__((ext_vector_type(16))) __bf16 v16bf;
typedef __attribute__((ext_vector_type(8)))  __bf16 v8bf;
typedef __attribute__((ext_vector_type(4)))  __bf16 v4bf;
typedef __attribute__((ext_vector_type(8)))  float  v8f;
typedef __attribute__((ext_vector_type(4)))  int    v4i;

#define AS1 __attribute__((address_space(1)))
#define AS3 __attribute__((address_space(3)))

static __device__ inline __bf16 tobf(float f) { return (__bf16)f; }

static __device__ inline v16bf cat8(v8bf lo, v8bf hi) {
    return __builtin_shufflevector(lo, hi, 0, 1, 2, 3, 4, 5, 6, 7,
                                   8, 9, 10, 11, 12, 13, 14, 15);
}

static __device__ inline v8f wmma_bf16(v16bf a, v16bf b, v8f c) {
    // D = A(16x32) * B(32x16) + C(16x16), f32 accum
    return __builtin_amdgcn_wmma_f32_16x16x32_bf16(
        /*neg_a=*/false, a, /*neg_b=*/false, b,
        /*c_mod=*/(short)0, c, /*reuse_a=*/false, /*reuse_b=*/false);
}

// async 16B global -> LDS copy (gfx1250 GLOBAL_LOAD_ASYNC_TO_LDS_B128)
static __device__ inline void async_cp16(const void* g, void* l) {
    __builtin_amdgcn_global_load_async_to_lds_b128(
        (AS1 v4i*)g, (AS3 v4i*)l, 0, 0);
}

static __device__ inline void wait_async0() {
    asm volatile("s_wait_asynccnt 0x0" ::: "memory");
}

// A-fragment load: contiguous K order (B uses the same), M = lane&15, K = 16g+e.
// 16 contiguous K values -> two 16B loads.
static __device__ inline v16bf load_afrag(const __bf16* row, int kk, int g) {
    v8bf lo = *(const v8bf*)(row + kk + 16 * g);
    v8bf hi = *(const v8bf*)(row + kk + 16 * g + 8);
    return cat8(lo, hi);
}

// B-fragment load: N = lane&15, K = 16g+e -> 16 contiguous values.
static __device__ inline v16bf load_bfrag(const __bf16* p) {
    v8bf lo = *(const v8bf*)p;
    v8bf hi = *(const v8bf*)(p + 8);
    return cat8(lo, hi);
}

// ---------------------------------------------------------------------------
// Kernel 0: convert an f32 weight matrix to bf16 (one-time, 64 blocks)
// ---------------------------------------------------------------------------
__global__ __launch_bounds__(256) void w2bf_kernel(
    const float* __restrict__ src, __bf16* __restrict__ dst, int n)
{
    int idx = (blockIdx.x * 256 + threadIdx.x) * 4;
    if (idx < n) {
        float4 f = *(const float4*)(src + idx);
        v4bf o = { tobf(f.x), tobf(f.y), tobf(f.z), tobf(f.w) };
        *(v4bf*)(dst + idx) = o;
    }
}

// ---------------------------------------------------------------------------
// Kernel 1: fused QKV projection.  grid = TOK/16 token tiles, 8 waves/block.
// Each wave computes two 16-wide output-channel tiles for q, k, v.
// ---------------------------------------------------------------------------
__global__ __launch_bounds__(256) void qkv_kernel(
    const float* __restrict__ x,
    const __bf16* __restrict__ Wq, const float* __restrict__ bq,
    const __bf16* __restrict__ Wk, const float* __restrict__ bk,
    const __bf16* __restrict__ Wv, const float* __restrict__ bv,
    float* __restrict__ q, float* __restrict__ k, float* __restrict__ v)
{
    __shared__ float  xf[16][CC];          // 16 KB raw f32 stage (async DMA)
    __shared__ __bf16 xs[16][CC];          // 8 KB bf16 operand tile
    const int tok0 = blockIdx.x * 16;
    const int tid  = threadIdx.x;

    // async-stage the token tile into LDS (ASYNCcnt), then convert to bf16
    for (int i = tid; i < 16 * CC / 4; i += 256) {
        int m = i >> 6, c4 = (i & 63) * 4;
        async_cp16(x + (size_t)(tok0 + m) * CC + c4, &xf[m][c4]);
    }
    wait_async0();
    __syncthreads();
    for (int i = tid; i < 16 * CC / 4; i += 256) {
        int m = i >> 6, c4 = (i & 63) * 4;
        float4 f = *(const float4*)&xf[m][c4];
        v4bf o = { tobf(f.x), tobf(f.y), tobf(f.z), tobf(f.w) };
        *(v4bf*)&xs[m][c4] = o;
    }
    __syncthreads();

    const int lane = tid & 31, wave = tid >> 5;
    const int g = lane >> 4, n = lane & 15;
    const int o0 = wave * 32;              // this wave's channel tiles: o0, o0+16

    v8f aq0 = {}, aq1 = {}, ak0 = {}, ak1 = {}, av0 = {}, av1 = {};

    for (int kk = 0; kk < CC; kk += 32) {
        v16bf a = load_afrag(xs[n], kk, g);
        const int kb = kk + g * 16;        // B layout: N = lane&15, K = 16g+e
        const __bf16* pq0 = Wq + (size_t)(o0 + n) * CC + kb;
        const __bf16* pq1 = Wq + (size_t)(o0 + 16 + n) * CC + kb;
        const __bf16* pk0 = Wk + (size_t)(o0 + n) * CC + kb;
        const __bf16* pk1 = Wk + (size_t)(o0 + 16 + n) * CC + kb;
        const __bf16* pv0 = Wv + (size_t)(o0 + n) * CC + kb;
        const __bf16* pv1 = Wv + (size_t)(o0 + 16 + n) * CC + kb;
        __builtin_prefetch(pq0 + 32, 0, 0);   // global_prefetch next K-slab
        aq0 = wmma_bf16(a, load_bfrag(pq0), aq0);
        aq1 = wmma_bf16(a, load_bfrag(pq1), aq1);
        ak0 = wmma_bf16(a, load_bfrag(pk0), ak0);
        ak1 = wmma_bf16(a, load_bfrag(pk1), ak1);
        av0 = wmma_bf16(a, load_bfrag(pv0), av0);
        av1 = wmma_bf16(a, load_bfrag(pv1), av1);
    }

    const float scaling = 0.17677669529663687f;   // dk^-0.5, dk=32
    const int c0 = o0 + n, c1 = o0 + 16 + n;
    const float bq0 = bq[c0], bq1 = bq[c1];
    const float bk0 = bk[c0], bk1 = bk[c1];
    const float bv0 = bv[c0], bv1 = bv[c1];
    #pragma unroll
    for (int r = 0; r < 8; ++r) {                 // C layout: M = r + 8g
        size_t base = (size_t)(tok0 + r + 8 * g) * CC;
        q[base + c0] = aq0[r] + bq0;
        q[base + c1] = aq1[r] + bq1;
        k[base + c0] = (ak0[r] + bk0) * scaling;
        k[base + c1] = (ak1[r] + bk1) * scaling;
        v[base + c0] = av0[r] + bv0;
        v[base + c1] = av1[r] + bv1;
    }
}

// ---------------------------------------------------------------------------
// Kernel 2/3: axial attention (templated on axis).
// One block per (b, line, head); 4 waves = 4 M-tiles of 16 rows (56 -> 64 pad).
// ---------------------------------------------------------------------------
template <bool WIDTH>
__global__ __launch_bounds__(128) void attn_kernel(
    const float* __restrict__ q, const float* __restrict__ k,
    const float* __restrict__ val, const float* __restrict__ mask,
    float* __restrict__ out)
{
    __shared__ float  Qf[56][DK];        // 7 KB each: raw f32 async-DMA stage
    __shared__ float  Kf[56][DK];
    __shared__ float  Vf[56][DK];
    __shared__ __bf16 Qs[64][DK];        // 4 KB
    __shared__ __bf16 Ks[64][DK];        // 4 KB
    __shared__ __bf16 VsT[DK][64];       // 4 KB, transposed: VsT[d][row]
    __shared__ __bf16 Ps[4][16][64];     // 8 KB: per-wave probability tiles

    const int bid  = blockIdx.x;
    const int head = bid & 7;
    const int line = (bid >> 3) % 56;
    const int b    = (bid >> 3) / 56;

    size_t base; int rowStride;
    if (WIDTH) { base = ((size_t)(b * CH + line) * CW) * CC + head * DK; rowStride = CC; }
    else       { base = ((size_t)(b * CH) * CW + line) * CC + head * DK; rowStride = CW * CC; }

    // 1) async-stage the 56x32 f32 q/k/v panels into LDS (ASYNCcnt path)
    for (int i = threadIdx.x; i < 56 * (DK / 4); i += 128) {
        int r = i >> 3, d4 = (i & 7) * 4;
        size_t a = base + (size_t)r * rowStride + d4;
        async_cp16(q + a, &Qf[r][d4]);
        async_cp16(k + a, &Kf[r][d4]);
        async_cp16(val + a, &Vf[r][d4]);
    }
    wait_async0();
    __syncthreads();

    // 2) convert to bf16 operand tiles (rows >= 56 zero-padded)
    for (int i = threadIdx.x; i < 64 * (DK / 4); i += 128) {
        int r = i >> 3, d4 = (i & 7) * 4;
        v4bf qv = {}, kv = {};
        __bf16 v0 = tobf(0.f), v1 = v0, v2 = v0, v3 = v0;
        if (r < 56) {
            float4 fq = *(const float4*)&Qf[r][d4];
            float4 fk = *(const float4*)&Kf[r][d4];
            float4 fv = *(const float4*)&Vf[r][d4];
            qv = v4bf{ tobf(fq.x), tobf(fq.y), tobf(fq.z), tobf(fq.w) };
            kv = v4bf{ tobf(fk.x), tobf(fk.y), tobf(fk.z), tobf(fk.w) };
            v0 = tobf(fv.x); v1 = tobf(fv.y); v2 = tobf(fv.z); v3 = tobf(fv.w);
        }
        *(v4bf*)&Qs[r][d4] = qv;
        *(v4bf*)&Ks[r][d4] = kv;
        VsT[d4 + 0][r] = v0; VsT[d4 + 1][r] = v1;
        VsT[d4 + 2][r] = v2; VsT[d4 + 3][r] = v3;
    }
    __syncthreads();

    const int lane = threadIdx.x & 31, wave = threadIdx.x >> 5;
    const int g = lane >> 4, n = lane & 15;
    const int rbase = wave * 16;

    // --- scores: S[16][64] = Q_tile (16x32) * K^T, 4 N-tiles of 16 ---
    v16bf aq = load_afrag(Qs[rbase + n], 0, g);
    v8f s[4] = {};
    #pragma unroll
    for (int j = 0; j < 4; ++j) {
        // B[k][nn] = Ks[j*16+nn][k]; K run 16g..16g+15 contiguous in row
        s[j] = wmma_bf16(aq, load_bfrag(&Ks[j * 16 + n][16 * g]), s[j]);
    }

    // --- mask add + row softmax (row = r + 8g, cols across lanes 0..15) ---
    const float NEG = -1e30f;
    #pragma unroll
    for (int r = 0; r < 8; ++r) {
        int i = rbase + r + 8 * g;
        float m = NEG;
        #pragma unroll
        for (int j = 0; j < 4; ++j) {
            int col = j * 16 + n;
            float sv = s[j][r];
            if (i < 56 && col < 56)
                sv += mask[((size_t)head * 56 + i) * 56 + col];
            else
                sv = NEG;
            s[j][r] = sv;
            m = fmaxf(m, sv);
        }
        #pragma unroll
        for (int off = 1; off < 16; off <<= 1) m = fmaxf(m, __shfl_xor(m, off, 16));
        float sum = 0.f;
        #pragma unroll
        for (int j = 0; j < 4; ++j) {
            float e_ = __expf(s[j][r] - m);
            s[j][r] = e_;
            sum += e_;
        }
        #pragma unroll
        for (int off = 1; off < 16; off <<= 1) sum += __shfl_xor(sum, off, 16);
        float inv = 1.0f / sum;
        #pragma unroll
        for (int j = 0; j < 4; ++j) s[j][r] *= inv;
    }

    // spill probabilities to LDS (C layout -> memory) and reload in A layout
    #pragma unroll
    for (int r = 0; r < 8; ++r)
        #pragma unroll
        for (int j = 0; j < 4; ++j)
            Ps[wave][r + 8 * g][j * 16 + n] = tobf(s[j][r]);
    __syncthreads();

    // --- output: O[16][32] = P (16x64) * V (64x32), 2 K-steps x 2 N-tiles ---
    v8f o0 = {}, o1 = {};
    #pragma unroll
    for (int kk = 0; kk < 64; kk += 32) {
        v16bf pa = load_afrag(Ps[wave][n], kk, g);
        // B[k][nn] = V[k][nn] = VsT[nn][k]; K run kk+16g..+15 contiguous
        o0 = wmma_bf16(pa, load_bfrag(&VsT[n][kk + 16 * g]), o0);
        o1 = wmma_bf16(pa, load_bfrag(&VsT[16 + n][kk + 16 * g]), o1);
    }

    #pragma unroll
    for (int r = 0; r < 8; ++r) {
        int i = rbase + r + 8 * g;
        if (i < 56) {
            size_t a = base + (size_t)i * rowStride;
            out[a + n]      = o0[r];
            out[a + 16 + n] = o1[r];
        }
    }
}

// ---------------------------------------------------------------------------
// Kernel 4: lepe depthwise 5x5 conv on v, accumulated into height-attn output
// ---------------------------------------------------------------------------
__global__ __launch_bounds__(256) void lepe_add_kernel(
    const float* __restrict__ v, const float* __restrict__ w,
    const float* __restrict__ bias, float* __restrict__ out2)
{
    size_t idx = (size_t)blockIdx.x * 256 + threadIdx.x;
    if (idx >= (size_t)TOK * CC) return;
    int ch = idx & 255;
    int t  = (int)(idx >> 8);
    int ww = t % CW;
    int hh = (t / CW) % CH;
    int b  = t / (CW * CH);
    float acc = bias[ch];
    #pragma unroll
    for (int dy = 0; dy < 5; ++dy) {
        int y = hh + dy - 2;
        if (y < 0 || y >= CH) continue;
        #pragma unroll
        for (int dx = 0; dx < 5; ++dx) {
            int xx = ww + dx - 2;
            if (xx < 0 || xx >= CW) continue;
            acc += v[(((size_t)(b * CH + y) * CW + xx) << 8) + ch] *
                   w[(size_t)(dy * 5 + dx) * CC + ch];
        }
    }
    out2[idx] += acc;
}

// ---------------------------------------------------------------------------
// Kernel 5: output projection out = in @ Wo^T + bo (same WMMA scheme as qkv)
// ---------------------------------------------------------------------------
__global__ __launch_bounds__(256) void proj_kernel(
    const float* __restrict__ in, const __bf16* __restrict__ Wo,
    const float* __restrict__ bo, float* __restrict__ out)
{
    __shared__ float  xf[16][CC];
    __shared__ __bf16 xs[16][CC];
    const int tok0 = blockIdx.x * 16;
    const int tid  = threadIdx.x;

    for (int i = tid; i < 16 * CC / 4; i += 256) {
        int m = i >> 6, c4 = (i & 63) * 4;
        async_cp16(in + (size_t)(tok0 + m) * CC + c4, &xf[m][c4]);
    }
    wait_async0();
    __syncthreads();
    for (int i = tid; i < 16 * CC / 4; i += 256) {
        int m = i >> 6, c4 = (i & 63) * 4;
        float4 f = *(const float4*)&xf[m][c4];
        v4bf o = { tobf(f.x), tobf(f.y), tobf(f.z), tobf(f.w) };
        *(v4bf*)&xs[m][c4] = o;
    }
    __syncthreads();

    const int lane = tid & 31, wave = tid >> 5;
    const int g = lane >> 4, n = lane & 15;
    const int o0 = wave * 32;

    v8f a0 = {}, a1 = {};
    for (int kk = 0; kk < CC; kk += 32) {
        v16bf a = load_afrag(xs[n], kk, g);
        const int kb = kk + g * 16;
        const __bf16* p0 = Wo + (size_t)(o0 + n) * CC + kb;
        const __bf16* p1 = Wo + (size_t)(o0 + 16 + n) * CC + kb;
        __builtin_prefetch(p0 + 32, 0, 0);
        a0 = wmma_bf16(a, load_bfrag(p0), a0);
        a1 = wmma_bf16(a, load_bfrag(p1), a1);
    }

    const int c0 = o0 + n, c1 = o0 + 16 + n;
    const float bb0 = bo[c0], bb1 = bo[c1];
    #pragma unroll
    for (int r = 0; r < 8; ++r) {
        size_t base = (size_t)(tok0 + r + 8 * g) * CC;
        out[base + c0] = a0[r] + bb0;
        out[base + c1] = a1[r] + bb1;
    }
}

// ---------------------------------------------------------------------------
extern "C" void kernel_launch(void* const* d_in, const int* in_sizes, int n_in,
                              void* d_out, int out_size, void* d_ws, size_t ws_size,
                              hipStream_t stream)
{
    const float* x      = (const float*)d_in[0];
    const float* mask_h = (const float*)d_in[1];
    const float* mask_w = (const float*)d_in[2];
    const float* Wq     = (const float*)d_in[3];
    const float* bq     = (const float*)d_in[4];
    const float* Wk     = (const float*)d_in[5];
    const float* bk     = (const float*)d_in[6];
    const float* Wv     = (const float*)d_in[7];
    const float* bv     = (const float*)d_in[8];
    const float* lepe_w = (const float*)d_in[9];
    const float* lepe_b = (const float*)d_in[10];
    const float* Wo     = (const float*)d_in[11];
    const float* bo     = (const float*)d_in[12];
    float* out = (float*)d_out;

    const size_t S = (size_t)TOK * CC;   // elements per (b,h,w,c) tensor
    const int    WN = CC * CC;           // 65536 weights per matrix
    float* ws  = (float*)d_ws;           // needs 5*S*4 + 4*WN*2 ~= 257.5 MB
    float* q   = ws;
    float* k   = ws + S;
    float* v   = ws + 2 * S;
    float* vv  = ws + 3 * S;
    float* o2  = ws + 4 * S;
    __bf16* wb = (__bf16*)(ws + 5 * S);  // bf16 weights: Wq,Wk,Wv,Wo
    __bf16* Wqb = wb;
    __bf16* Wkb = wb + WN;
    __bf16* Wvb = wb + 2 * WN;
    __bf16* Wob = wb + 3 * WN;

    // 0) one-time weight conversion to bf16
    w2bf_kernel<<<WN / 4 / 256, 256, 0, stream>>>(Wq, Wqb, WN);
    w2bf_kernel<<<WN / 4 / 256, 256, 0, stream>>>(Wk, Wkb, WN);
    w2bf_kernel<<<WN / 4 / 256, 256, 0, stream>>>(Wv, Wvb, WN);
    w2bf_kernel<<<WN / 4 / 256, 256, 0, stream>>>(Wo, Wob, WN);

    // 1) fused QKV projection (WMMA bf16)
    qkv_kernel<<<TOK / 16, 256, 0, stream>>>(x, Wqb, bq, Wkb, bk, Wvb, bv, q, k, v);

    // 2) width-axis attention: values = v, mask = mask_w -> vv
    attn_kernel<true><<<CB * CH * NH, 128, 0, stream>>>(q, k, v, mask_w, vv);

    // 3) height-axis attention: values = vv, mask = mask_h -> o2
    attn_kernel<false><<<CB * CW * NH, 128, 0, stream>>>(q, k, vv, mask_h, o2);

    // 4) lepe depthwise 5x5 conv on v, added into o2
    lepe_add_kernel<<<(unsigned)((S + 255) / 256), 256, 0, stream>>>(v, lepe_w, lepe_b, o2);

    // 5) output projection -> d_out
    proj_kernel<<<TOK / 16, 256, 0, stream>>>(o2, Wob, bo, out);
}
